// FFE1D_57140244906030
// MI455X (gfx1250) — matrix operand order repaired and checked by reading, not compiled, not run on hardware
//
#include <hip/hip_runtime.h>
#include <hip/hip_bf16.h>

typedef __attribute__((ext_vector_type(16))) _Float16 v16h;
typedef __attribute__((ext_vector_type(8)))  float    v8f;

#define BB 128
#define CC 320
#define NP 256      // 16x16 pixels
#define KW 2880     // 9*320 implicit-GEMM K
#define LDSW 328    // padded f16 row stride in LDS (320 + 8)

union FragH {
  v16h v;
  uint4 q[2];
  unsigned u[8];
  _Float16 h[16];
};

// A-matrix fragment (16x32 f16), row-major A[row][k], contiguous K order (B uses the same):
// lanes 0-15: M=lane, K = 0..15; lanes 16-31: M=lane-16, K = 16..31
__device__ __forceinline__ v16h frag_a_g(const _Float16* __restrict__ A, int row,
                                         int k0, int lda, int lane) {
  const _Float16* p = A + (size_t)row * lda + k0 + ((lane >> 4) << 4);
  FragH f;
  f.q[0] = *(const uint4*)p;         // K +0..7
  f.q[1] = *(const uint4*)(p + 8);   // K +8..15
  return f.v;
}

// B-matrix fragment (32x16 f16) from K-transposed storage Bt[n][k]:
// lanes 0-15: N=lane, K=0..15 (pairs per VGPR); lanes 16-31: N=lane-16, K=16..31
__device__ __forceinline__ v16h frag_b_gT(const _Float16* __restrict__ Bt, int col,
                                          int k0, int ldb, int lane) {
  const _Float16* p = Bt + (size_t)col * ldb + k0 + ((lane >> 4) << 4);
  FragH f;
  f.q[0] = *(const uint4*)p;
  f.q[1] = *(const uint4*)(p + 8);
  return f.v;
}

// ---------------------------------------------------------------------------
// k_masks: build the three K-transposed 256x256 f16 band-filter matrices.
// band(x) = x (circ-conv) h, h = IFFT2(Msym), Msym(u) = (M(u)+M(-u))/2.
// Ht[band][p][q] = h[(p0-q0)&15][(p1-q1)&15]
// ---------------------------------------------------------------------------
__global__ __launch_bounds__(256) void k_masks(const float* __restrict__ lcp,
                                               const float* __restrict__ hcp,
                                               _Float16* __restrict__ Ht) {
  __shared__ float msk[3][256];
  __shared__ float hh[3][256];
  const int t = threadIdx.x;
  const float lc = *lcp, hc = *hcp;
  {
    const int sy = t >> 4, sx = t & 15;
    const float yy = -1.f + (2.f / 15.f) * (float)sy;
    const float xx = -1.f + (2.f / 15.f) * (float)sx;
    const float d = sqrtf(xx * xx + yy * yy) * 0.70710678f;
    const float sl  = 1.f / (1.f + expf(-(lc - d) * 1e6f));
    const float sm1 = 1.f / (1.f + expf(-(d - lc) * 1e6f));
    const float sm2 = 1.f / (1.f + expf(-(hc - d) * 1e6f));
    const float sh  = 1.f / (1.f + expf(-(d - hc) * 1e6f));
    msk[0][t] = sl; msk[1][t] = sm1 * sm2; msk[2][t] = sh;
  }
  __syncthreads();
  {
    const int n0 = t >> 4, n1 = t & 15;
    float a0 = 0.f, a1 = 0.f, a2 = 0.f;
    for (int u = 0; u < 256; ++u) {
      const int u0 = u >> 4, u1 = u & 15;
      const int s  = ((((u0 + 8) & 15) << 4) | ((u1 + 8) & 15));
      const int u0n = (16 - u0) & 15, u1n = (16 - u1) & 15;
      const int sn = ((((u0n + 8) & 15) << 4) | ((u1n + 8) & 15));
      const float c = cosf((float)((u0 * n0 + u1 * n1) & 15) * (6.28318530718f / 16.f));
      a0 += 0.5f * (msk[0][s] + msk[0][sn]) * c;
      a1 += 0.5f * (msk[1][s] + msk[1][sn]) * c;
      a2 += 0.5f * (msk[2][s] + msk[2][sn]) * c;
    }
    hh[0][t] = a0 * (1.f / 256.f);
    hh[1][t] = a1 * (1.f / 256.f);
    hh[2][t] = a2 * (1.f / 256.f);
  }
  __syncthreads();
  for (int w = t; w < 3 * 65536; w += 256) {
    const int band = w >> 16;
    const int pq = w & 65535;
    const int p = pq >> 8, q = pq & 255;
    const int d0 = ((p >> 4) - (q >> 4)) & 15;
    const int d1 = ((p & 15) - (q & 15)) & 15;
    Ht[w] = (_Float16)hh[band][(d0 << 4) | d1];
  }
}

// ---------------------------------------------------------------------------
// k_cvt: x f32 -> f16 (A-matrix of the band GEMM), [b][c][p] row-major.
// ---------------------------------------------------------------------------
__global__ __launch_bounds__(256) void k_cvt(const float* __restrict__ x,
                                             _Float16* __restrict__ xh) {
  const size_t i = ((size_t)blockIdx.x * 256 + threadIdx.x) * 4;
  const float4 v = *(const float4*)(x + i);
  _Float16 h4[4] = {(_Float16)v.x, (_Float16)v.y, (_Float16)v.z, (_Float16)v.w};
  *(uint2*)(xh + i) = *(const uint2*)h4;
}

// ---------------------------------------------------------------------------
// k_repack: conv weights (C,C,3,3) f32 -> Apack[band][co][sh*320+ci] f16,
// plus GN params gnp[band][bias|gamma|beta][320].
// ---------------------------------------------------------------------------
__global__ __launch_bounds__(256) void k_repack(
    const float* __restrict__ lw, const float* __restrict__ mw, const float* __restrict__ hw,
    const float* __restrict__ lb, const float* __restrict__ lg, const float* __restrict__ lbe,
    const float* __restrict__ mb, const float* __restrict__ mg, const float* __restrict__ mbe,
    const float* __restrict__ hb, const float* __restrict__ hg, const float* __restrict__ hbe,
    _Float16* __restrict__ Apack, float* __restrict__ gnp) {
  const int i = blockIdx.x * 256 + threadIdx.x;
  if (i < CC * KW) {
    const int co = i / KW;
    const int k = i - co * KW;
    const int sh = k / CC;
    const int ci = k - sh * CC;
    const int src = (co * CC + ci) * 9 + sh;
    Apack[i]               = (_Float16)lw[src];
    Apack[CC * KW + i]     = (_Float16)mw[src];
    Apack[2 * CC * KW + i] = (_Float16)hw[src];
  }
  if (i < CC) {
    gnp[i] = lb[i];        gnp[320 + i] = lg[i];  gnp[640 + i] = lbe[i];
    gnp[960 + i] = mb[i];  gnp[1280 + i] = mg[i]; gnp[1600 + i] = mbe[i];
    gnp[1920 + i] = hb[i]; gnp[2240 + i] = hg[i]; gnp[2560 + i] = hbe[i];
  }
}

// ---------------------------------------------------------------------------
// k_ca: channel attention -> cw[b][3][320]
// ---------------------------------------------------------------------------
__global__ __launch_bounds__(320) void k_ca(const float* __restrict__ x,
                                            const float* __restrict__ w1,
                                            const float* __restrict__ b1,
                                            const float* __restrict__ w2,
                                            const float* __restrict__ b2,
                                            float* __restrict__ cw) {
  __shared__ float avg[CC];
  __shared__ float hid[40];
  const int b = blockIdx.x, t = threadIdx.x;
  const float* row = x + ((size_t)b * CC + t) * NP;
  float s = 0.f;
  for (int n = 0; n < NP; ++n) s += row[n];
  avg[t] = s * (1.f / 256.f);
  __syncthreads();
  if (t < 40) {
    float a = b1[t];
    const float* wr = w1 + t * CC;
    for (int c = 0; c < CC; ++c) a += avg[c] * wr[c];
    hid[t] = fmaxf(a, 0.f);
  }
  __syncthreads();
  for (int i = 0; i < 3; ++i) {
    const int r = i * CC + t;
    float a = b2[r];
    const float* wr = w2 + (size_t)r * 40;
    for (int k = 0; k < 40; ++k) a += hid[k] * wr[k];
    cw[((size_t)b * 3 + i) * CC + t] = 1.f / (1.f + expf(-a));
  }
}

// ---------------------------------------------------------------------------
// k_band: WMMA GEMM  Y = xh(40960x256) * Ht^T(256x256) per band.
// Output layout Y[band][b][p][c] f16.
// ---------------------------------------------------------------------------
__global__ __launch_bounds__(256) void k_band(const _Float16* __restrict__ xh,
                                              const _Float16* __restrict__ Htm,
                                              _Float16* __restrict__ Y) {
  const int band = blockIdx.y;
  const int mbk = blockIdx.x;                 // 0..639
  const int tid = threadIdx.x;
  const int wave = tid >> 5, lane = tid & 31;
  const int wm = wave & 3;                    // m-tile in block
  const int wn0 = (wave >> 2) << 3;           // first n-tile (0 or 8)
  const int row = mbk * 64 + wm * 16 + (lane & 15);
  const _Float16* Bt = Htm + (size_t)band * 65536;

  v8f acc[8] = {};
  for (int kt = 0; kt < 8; ++kt) {
    const int k0 = kt * 32;
    if (kt < 7) __builtin_prefetch(xh + (size_t)row * NP + k0 + 32, 0, 1);
    const v16h a = frag_a_g(xh, row, k0, NP, lane);
#pragma unroll
    for (int nt = 0; nt < 8; ++nt) {
      const int col = (wn0 + nt) * 16 + (lane & 15);
      const v16h bf = frag_b_gT(Bt, col, k0, NP, lane);
      acc[nt] = __builtin_amdgcn_wmma_f32_16x16x32_f16(false, a, false, bf,
                                                       (short)0, acc[nt], false, false);
    }
  }
  const int r8 = (lane >> 4) << 3;
#pragma unroll
  for (int nt = 0; nt < 8; ++nt) {
    const int p = (wn0 + nt) * 16 + (lane & 15);
#pragma unroll
    for (int r = 0; r < 8; ++r) {
      const int bc = mbk * 64 + wm * 16 + r8 + r;
      const int bb = bc / CC, c = bc - bb * CC;
      Y[((size_t)(band * BB + bb) * NP + p) * CC + c] = (_Float16)acc[nt][r];
    }
  }
}

// ---------------------------------------------------------------------------
// k_conv_gn: per (group, batch, band) block.
// Implicit-GEMM 3x3 conv (64 co x 256 px, K = 9*320) with WMMA f16,
// fused bias + GroupNorm (block == exactly one group) + affine + residual.
// Input tile (256 px x 320 ci f16) staged once into LDS (164 KB dynamic)
// via CDNA5 async global->LDS DMA (ASYNCcnt-tracked, no VGPR round trip).
// ---------------------------------------------------------------------------
__global__ __launch_bounds__(256) void k_conv_gn(const _Float16* __restrict__ Y,
                                                 const _Float16* __restrict__ Apack,
                                                 const float* __restrict__ gnp,
                                                 const float* __restrict__ x,
                                                 _Float16* __restrict__ feat) {
  extern __shared__ uint4 lin4[];              // [256][LDSW] f16, 16B aligned
  _Float16* lin = (_Float16*)lin4;
  __shared__ float reds[16];
  __shared__ float mu_s, rs_s;

  const int g = blockIdx.x;                    // group 0..4 (== 64-row M block)
  const int b = blockIdx.y;
  const int band = blockIdx.z;
  const int tid = threadIdx.x;
  const int wave = tid >> 5, lane = tid & 31;
  const int wm = wave & 3;
  const int wn0 = (wave >> 2) << 3;

  // stage band input [p][c] -> padded LDS rows with async global->LDS b128 DMA.
  // AS3 LDS offset = low 32 bits of the flat aperture address of `lin`.
  {
    const char* src = (const char*)(Y + (size_t)(band * BB + b) * NP * CC);
    const unsigned lbase = (unsigned)(size_t)lin;
    for (int w = tid; w < NP * 40; w += 256) {          // 40 x 16B chunks per row
      const int p = w / 40, c16 = w - p * 40;
      const unsigned ldst = lbase + (unsigned)(p * (LDSW * 2) + c16 * 16);
      const unsigned long long ga = (unsigned long long)src + (unsigned long long)w * 16ull;
      asm volatile("global_load_async_to_lds_b128 %0, %1, off"
                   :: "v"(ldst), "v"(ga) : "memory");
    }
#if __has_builtin(__builtin_amdgcn_s_wait_asynccnt)
    __builtin_amdgcn_s_wait_asynccnt(0);
#else
    asm volatile("s_wait_asynccnt 0x0" ::: "memory");
#endif
  }
  __syncthreads();

  const _Float16* Ab = Apack + (size_t)band * CC * KW;
  const int row = g * 64 + wm * 16 + (lane & 15);
  const int lp = (lane & 15);

  v8f acc[8] = {};
  for (int sh = 0; sh < 9; ++sh) {
    const int dy = sh / 3 - 1, dx = sh % 3 - 1;
    int rowoff[8];
    bool okk[8];
#pragma unroll
    for (int nt = 0; nt < 8; ++nt) {
      const int p = (wn0 + nt) * 16 + lp;
      const int sy = (p >> 4) + dy, sx = (p & 15) + dx;
      const bool ok = ((unsigned)sy < 16u) && ((unsigned)sx < 16u);
      okk[nt] = ok;
      rowoff[nt] = ok ? (sy * 16 + sx) * LDSW : 0;     // clamped row, branchless
    }
    for (int kt = 0; kt < 10; ++kt) {
      const int k0 = sh * CC + kt * 32;
      const v16h a = frag_a_g(Ab, row, k0, KW, lane);
      const int kc = kt * 32 + ((lane >> 4) << 4);     // ci base of this lane's K half
#pragma unroll
      for (int nt = 0; nt < 8; ++nt) {
        FragH f;
        const _Float16* lp16 = lin + rowoff[nt] + kc;
        f.q[0] = *(const uint4*)lp16;
        f.q[1] = *(const uint4*)(lp16 + 8);
#pragma unroll
        for (int j = 0; j < 8; ++j) f.u[j] = okk[nt] ? f.u[j] : 0u;  // v_cndmask
        acc[nt] = __builtin_amdgcn_wmma_f32_16x16x32_f16(false, a, false, f.v,
                                                         (short)0, acc[nt], false, false);
      }
    }
  }

  // bias + GroupNorm statistics over this block's 64ch x 256px = one group
  const float* gp = gnp + band * 960;
  const int r8 = (lane >> 4) << 3;
  const int coBase = g * 64 + wm * 16 + r8;
  float brow[8];
#pragma unroll
  for (int r = 0; r < 8; ++r) brow[r] = gp[coBase + r];

  float s = 0.f, s2 = 0.f;
#pragma unroll
  for (int nt = 0; nt < 8; ++nt)
#pragma unroll
    for (int r = 0; r < 8; ++r) {
      const float v = acc[nt][r] + brow[r];
      acc[nt][r] = v;
      s += v; s2 += v * v;
    }
#pragma unroll
  for (int off = 16; off > 0; off >>= 1) {
    s  += __shfl_xor(s, off, 32);
    s2 += __shfl_xor(s2, off, 32);
  }
  if (lane == 0) { reds[wave] = s; reds[8 + wave] = s2; }
  __syncthreads();
  if (tid == 0) {
    float S = 0.f, S2 = 0.f;
    for (int w = 0; w < 8; ++w) { S += reds[w]; S2 += reds[8 + w]; }
    const float mu = S * (1.f / 16384.f);
    const float var = S2 * (1.f / 16384.f) - mu * mu;
    mu_s = mu;
    rs_s = rsqrtf(var + 1e-5f);
  }
  __syncthreads();
  const float mu = mu_s, rs = rs_s;

  float gam[8], bet[8];
#pragma unroll
  for (int r = 0; r < 8; ++r) { gam[r] = gp[320 + coBase + r]; bet[r] = gp[640 + coBase + r]; }

  const float* xr = x + ((size_t)b * CC + coBase) * NP;
  _Float16* fo = feat + ((size_t)(band * BB + b) * CC + coBase) * NP;
#pragma unroll
  for (int nt = 0; nt < 8; ++nt) {
    const int p = (wn0 + nt) * 16 + lp;
#pragma unroll
    for (int r = 0; r < 8; ++r) {
      const float v = (acc[nt][r] - mu) * rs * gam[r] + bet[r] + xr[(size_t)r * NP + p];
      fo[(size_t)r * NP + p] = (_Float16)v;
    }
  }
}

// ---------------------------------------------------------------------------
// k_fuse: fused = sum_i cw[b][i][c]*feat_i ; spatial attention ; f32 output.
// ---------------------------------------------------------------------------
__global__ __launch_bounds__(256) void k_fuse(const _Float16* __restrict__ feat,
                                              const float* __restrict__ cw,
                                              const float* __restrict__ sa_w,
                                              const float* __restrict__ sa_b,
                                              float* __restrict__ out) {
  __shared__ float fuse[CC * 33];
  __shared__ float part[256];
  __shared__ float swv[32];
  const int b = blockIdx.x, pt = blockIdx.y;
  const int t = threadIdx.x;
  const int pc = t & 31, cch = t >> 5;
  const int p = pt * 32 + pc;

  float ps = 0.f;
  for (int c = cch * 40; c < cch * 40 + 40; ++c) {
    float f = 0.f;
#pragma unroll
    for (int i = 0; i < 3; ++i) {
      const float w = cw[((size_t)b * 3 + i) * CC + c];
      f += w * (float)feat[((size_t)(i * BB + b) * CC + c) * NP + p];
    }
    fuse[c * 33 + pc] = f;
    ps += f * sa_w[c];
  }
  part[t] = ps;
  __syncthreads();
  if (t < 32) {
    float sv = 0.f;
    for (int j = 0; j < 8; ++j) sv += part[j * 32 + t];
    swv[t] = 1.f / (1.f + expf(-(sv + sa_b[0])));
  }
  __syncthreads();
  const float sw = swv[pc];
  for (int c = cch * 40; c < cch * 40 + 40; ++c)
    out[((size_t)b * CC + c) * NP + p] = fuse[c * 33 + pc] * sw;
}

// ---------------------------------------------------------------------------
extern "C" void kernel_launch(void* const* d_in, const int* in_sizes, int n_in,
                              void* d_out, int out_size, void* d_ws, size_t ws_size,
                              hipStream_t stream) {
  (void)in_sizes; (void)n_in; (void)out_size; (void)ws_size;
  const float* x     = (const float*)d_in[0];
  const float* lc    = (const float*)d_in[1];
  const float* hc    = (const float*)d_in[2];
  const float* ca_w1 = (const float*)d_in[3];
  const float* ca_b1 = (const float*)d_in[4];
  const float* ca_w2 = (const float*)d_in[5];
  const float* ca_b2 = (const float*)d_in[6];
  const float* low_w  = (const float*)d_in[7];
  const float* low_b  = (const float*)d_in[8];
  const float* low_g  = (const float*)d_in[9];
  const float* low_be = (const float*)d_in[10];
  const float* mid_w  = (const float*)d_in[11];
  const float* mid_b  = (const float*)d_in[12];
  const float* mid_g  = (const float*)d_in[13];
  const float* mid_be = (const float*)d_in[14];
  const float* high_w  = (const float*)d_in[15];
  const float* high_b  = (const float*)d_in[16];
  const float* high_g  = (const float*)d_in[17];
  const float* high_be = (const float*)d_in[18];
  const float* sa_w = (const float*)d_in[19];
  const float* sa_b = (const float*)d_in[20];

  char* ws = (char*)d_ws;
  _Float16* Ht    = (_Float16*)(ws + 0);          // 3*256*256*2      = 393,216
  _Float16* xh    = (_Float16*)(ws + 393216);     // 128*320*256*2    = 20,971,520
  _Float16* Apack = (_Float16*)(ws + 21364736);   // 3*320*2880*2     = 5,529,600
  float*    gnp   = (float*)   (ws + 26894336);   // 3*960*4          = 11,520
  float*    cwb   = (float*)   (ws + 26905856);   // 128*3*320*4      = 491,520
  _Float16* Y     = (_Float16*)(ws + 27397376);   // 3*128*256*320*2  = 62,914,560
  _Float16* feat  = (_Float16*)(ws + 90311936);   // 3*128*320*256*2  = 62,914,560

  k_masks<<<1, 256, 0, stream>>>(lc, hc, Ht);
  k_cvt<<<10240, 256, 0, stream>>>(x, xh);
  k_repack<<<3600, 256, 0, stream>>>(low_w, mid_w, high_w,
                                     low_b, low_g, low_be,
                                     mid_b, mid_g, mid_be,
                                     high_b, high_g, high_be,
                                     Apack, gnp);
  k_ca<<<128, 320, 0, stream>>>(x, ca_w1, ca_b1, ca_w2, ca_b2, cwb);
  k_band<<<dim3(640, 3), 256, 0, stream>>>(xh, Ht, Y);
  k_conv_gn<<<dim3(5, 128, 3), 256, NP * LDSW * sizeof(_Float16), stream>>>(Y, Apack, gnp, x, feat);
  k_fuse<<<dim3(128, 8), 256, 0, stream>>>(feat, cwb, sa_w, sa_b, (float*)d_out);
}
